// MambaAudioClassifier_15522011808171
// MI455X (gfx1250) — matrix operand order repaired and checked by reading, not compiled, not run on hardware
//
#include <hip/hip_runtime.h>
#include <hip/hip_bf16.h>
#include <cstddef>
#include <cstdint>

// ---------------------------------------------------------------------------
// Mamba audio classifier forward pass for gfx1250 (MI455X).
// GEMMs: v_wmma_f32_16x16x32_f16, LDS-staged with double-buffered
// global_load_async_to_lds_b128 (ASYNCcnt) issued via inline asm.
// ---------------------------------------------------------------------------

#define D_MODEL   512
#define D_INNER   1024
#define D_STATE   16
#define DT_RANK   32
#define D_CONV    4
#define N_LAYER   12
#define D_MLP     2048
#define N_MELS    128
#define N_CLASSES 50
#define BATCH     4
#define SEQ       512
#define MTOK      (BATCH * SEQ)   // 2048

typedef __attribute__((ext_vector_type(16))) _Float16 v16h;
typedef __attribute__((ext_vector_type(8)))  float    v8f;

// GEMM blocking: block = 256 threads = 8 waves (4 M-waves x 2 N-waves).
// Block tile 64(M) x 64(N); wave tile 16(M) x 32(N); K step 32.
#define BM 64
#define BN 64
#define BK 32
#define LDS_STRIDE 36   // 32 floats + 4 pad (keeps 16B chunks intact)

// -------------------- async global->LDS (gfx1250) helpers ------------------

#if defined(__gfx1250__)
#define HAVE_ASYNC_LDS 1
#else
#define HAVE_ASYNC_LDS 0
#endif

__device__ __forceinline__ void async_copy16(const float* __restrict__ g, float* l) {
#if HAVE_ASYNC_LDS
    // GLOBAL_LOAD_ASYNC_TO_LDS_B128: VDST = per-lane LDS byte offset,
    // VADDR = 64-bit global address, no SADDR ("off").  Tracked by ASYNCcnt.
    // Generic LDS pointers carry the LDS byte offset in their low 32 bits
    // (ISA 10.2 aperture mapping).
    const uint32_t lofs = (uint32_t)(uintptr_t)l;
    asm volatile("global_load_async_to_lds_b128 %0, %1, off"
                 :
                 : "v"(lofs), "v"((const void*)g)
                 : "memory");
#else
    *reinterpret_cast<float4*>(l) = *reinterpret_cast<const float4*>(g);
#endif
}

__device__ __forceinline__ void wait_async_all() {
#if HAVE_ASYNC_LDS
#if __has_builtin(__builtin_amdgcn_s_wait_asynccnt)
    __builtin_amdgcn_s_wait_asynccnt(0);
#else
    asm volatile("s_wait_asynccnt 0x0" ::: "memory");
#endif
#endif
}

// ----------------------------- device helpers ------------------------------

__device__ __forceinline__ float siluf(float x) {
    return x / (1.0f + __expf(-x));
}
__device__ __forceinline__ float softplusf(float x) {
    return (x > 20.0f) ? x : log1pf(__expf(x));
}

// Stage a 64-row x 32-col f32 slab (row-major, leading dim ldg) into LDS with
// row stride LDS_STRIDE. 64 rows x 8 chunks(16B) = 512 chunks over 256 thrds.
__device__ __forceinline__ void stage_tile(const float* __restrict__ G, int ldg,
                                           int row0, int k0, float* dst) {
    const int t = threadIdx.x;
#pragma unroll
    for (int c = 0; c < 2; ++c) {
        const int chunk = t + c * 256;       // 0..511
        const int r  = chunk >> 3;
        const int cc = chunk & 7;
        async_copy16(G + (size_t)(row0 + r) * ldg + k0 + cc * 4,
                     dst + r * LDS_STRIDE + cc * 4);
    }
}

__device__ __forceinline__ v16h pack16(float4 q0, float4 q1, float4 q2, float4 q3) {
    v16h f;
    f[0]  = (_Float16)q0.x; f[1]  = (_Float16)q0.y; f[2]  = (_Float16)q0.z; f[3]  = (_Float16)q0.w;
    f[4]  = (_Float16)q1.x; f[5]  = (_Float16)q1.y; f[6]  = (_Float16)q1.z; f[7]  = (_Float16)q1.w;
    f[8]  = (_Float16)q2.x; f[9]  = (_Float16)q2.y; f[10] = (_Float16)q2.z; f[11] = (_Float16)q2.w;
    f[12] = (_Float16)q3.x; f[13] = (_Float16)q3.y; f[14] = (_Float16)q3.z; f[15] = (_Float16)q3.w;
    return f;
}

// A-fragment 16x32 from LDS (s = base of this wave's 16-row slice).
// ISA 16-bit A 16x32 layout: lane half -> K offset 8 inside each 16-block.
__device__ __forceinline__ v16h lds_a_frag(const float* s) {
    const int lane = threadIdx.x & 31;
    const float* p = s + (lane & 15) * LDS_STRIDE + ((lane >> 4) << 3);
    return pack16(*reinterpret_cast<const float4*>(p),
                  *reinterpret_cast<const float4*>(p + 4),
                  *reinterpret_cast<const float4*>(p + 16),
                  *reinterpret_cast<const float4*>(p + 20));
}

// B-fragment (W rows = columns of C): same K layout as A (lane half -> K
// offset 8 inside each 16-block).
__device__ __forceinline__ v16h lds_b_frag(const float* s) {
    const int lane = threadIdx.x & 31;
    const float* p = s + (lane & 15) * LDS_STRIDE + ((lane >> 4) << 3);
    return pack16(*reinterpret_cast<const float4*>(p),
                  *reinterpret_cast<const float4*>(p + 4),
                  *reinterpret_cast<const float4*>(p + 16),
                  *reinterpret_cast<const float4*>(p + 20));
}

// Store one 16x16 f32 accumulator tile: lane 0-15 -> N=lane, M=vgpr;
// lane 16-31 -> N=lane-16, M=8+vgpr.
template <int ACT, bool BIAS>
__device__ __forceinline__ void store_tile(float* __restrict__ C, int ldc,
                                           int m0, int n0, v8f acc,
                                           const float* __restrict__ bias) {
    const int lane = threadIdx.x & 31;
    const int n  = n0 + (lane & 15);
    const int mb = m0 + ((lane >> 4) << 3);
    const float bv = BIAS ? bias[n] : 0.0f;
#pragma unroll
    for (int v = 0; v < 8; ++v) {
        float val = acc[v] + bv;
        if (ACT == 1) val = softplusf(val);
        C[(size_t)(mb + v) * ldc + n] = val;
    }
}

// --------------------------- WMMA GEMM kernel ------------------------------
// C[M,N] = act(A[M,K] @ W[N,K]^T + bias).  grid = (M/64, N/64), block = 256.
template <int ACT, bool BIAS>
__global__ __launch_bounds__(256) void wmma_gemm_kernel(
    const float* __restrict__ A, int lda,
    const float* __restrict__ W, int ldw,
    const float* __restrict__ bias,
    float* __restrict__ C, int ldc,
    int K) {
    __shared__ float sA[2][BM * LDS_STRIDE];
    __shared__ float sB[2][BN * LDS_STRIDE];

    const int m0   = blockIdx.x * BM;
    const int nblk = blockIdx.y * BN;
    const int wave = threadIdx.x >> 5;
    const int wm   = wave >> 1;    // 0..3 -> M sub-tile
    const int wn   = wave & 1;     // 0..1 -> N sub-tile

    v8f acc0 = {};
    v8f acc1 = {};

    stage_tile(A, lda, m0,   0, sA[0]);
    stage_tile(W, ldw, nblk, 0, sB[0]);

    const int nk = K / BK;
    for (int kb = 0; kb < nk; ++kb) {
        wait_async_all();        // this wave's stage(kb) copies have landed
        __syncthreads();         // ... and everyone else's

        if (kb + 1 < nk) {       // overlap next stage with current compute
            stage_tile(A, lda, m0,   (kb + 1) * BK, sA[(kb + 1) & 1]);
            stage_tile(W, ldw, nblk, (kb + 1) * BK, sB[(kb + 1) & 1]);
        }

        const float* a_base = sA[kb & 1] + (wm * 16) * LDS_STRIDE;
        const float* b_base = sB[kb & 1] + (wn * 32) * LDS_STRIDE;

        v16h a  = lds_a_frag(a_base);
        v16h b0 = lds_b_frag(b_base);
        acc0 = __builtin_amdgcn_wmma_f32_16x16x32_f16(
            false, a, false, b0, (short)0, acc0, false, false);
        v16h b1 = lds_b_frag(b_base + 16 * LDS_STRIDE);
        acc1 = __builtin_amdgcn_wmma_f32_16x16x32_f16(
            false, a, false, b1, (short)0, acc1, false, false);

        __syncthreads();         // done reading buf[kb&1] before it is reused
    }

    const int mrow = m0 + wm * 16;
    const int ncol = nblk + wn * 32;
    store_tile<ACT, BIAS>(C, ldc, mrow, ncol,      acc0, bias);
    store_tile<ACT, BIAS>(C, ldc, mrow, ncol + 16, acc1, bias);
}

// --------------------------- elementwise kernels ---------------------------

__global__ void zero_kernel(float* __restrict__ p, int n) {
    int i = blockIdx.x * blockDim.x + threadIdx.x;
    if (i < n) p[i] = 0.0f;
}

__global__ void add_kernel(float* __restrict__ acc, const float* __restrict__ x, int n) {
    int i = blockIdx.x * blockDim.x + threadIdx.x;
    if (i < n) acc[i] += x[i];
}

// RMSNorm over rows of length 512: one wave per row, 16 elems per lane.
__global__ __launch_bounds__(256) void rmsnorm_kernel(
    const float* __restrict__ x, const float* __restrict__ w,
    float* __restrict__ y, int rows) {
    const int wave = threadIdx.x >> 5;
    const int lane = threadIdx.x & 31;
    const int row = blockIdx.x * 8 + wave;
    if (row >= rows) return;
    const float* xr = x + (size_t)row * D_MODEL;
    float ss = 0.0f;
#pragma unroll
    for (int i = 0; i < D_MODEL / 32; ++i) {
        float v = xr[lane + 32 * i];
        ss += v * v;
    }
#pragma unroll
    for (int m = 16; m >= 1; m >>= 1) ss += __shfl_xor(ss, m, 32);
    const float scale = rsqrtf(ss * (1.0f / D_MODEL) + 1e-5f);
    float* yr = y + (size_t)row * D_MODEL;
#pragma unroll
    for (int i = 0; i < D_MODEL / 32; ++i) {
        int c = lane + 32 * i;
        yr[c] = xr[c] * scale * w[c];
    }
}

// Depthwise causal conv (width 4) + bias + SiLU. xs lives in the first
// D_INNER columns of xz (row stride 2*D_INNER).
__global__ void conv_silu_kernel(const float* __restrict__ xz,
                                 const float* __restrict__ cw,
                                 const float* __restrict__ cb,
                                 float* __restrict__ xc) {
    int i = blockIdx.x * blockDim.x + threadIdx.x;
    if (i >= MTOK * D_INNER) return;
    const int d  = i & (D_INNER - 1);
    const int rt = i >> 10;           // b*SEQ + t
    const int t  = rt & (SEQ - 1);
    const float* cwd = cw + d * D_CONV;
    float acc = cb[d];
#pragma unroll
    for (int j = 0; j < D_CONV; ++j) {
        int tt = t + j - (D_CONV - 1);
        if (tt >= 0)
            acc += cwd[j] * xz[(size_t)(rt + j - (D_CONV - 1)) * (2 * D_INNER) + d];
    }
    xc[i] = siluf(acc);
}

// Selective scan: one thread per (batch, channel); 16-state recurrence in
// registers, 512 sequential steps (v_exp_f32 heavy).
__global__ __launch_bounds__(256) void scan_kernel(
    const float* __restrict__ xdb,     // [MTOK, 64]: B at +32, C at +48
    const float* __restrict__ dt,      // [MTOK, D_INNER]
    const float* __restrict__ xc,      // [MTOK, D_INNER]
    const float* __restrict__ a_log,   // [D_INNER, 16] (layer slice)
    float* __restrict__ yscan) {       // [MTOK, D_INNER]
    int tid = blockIdx.x * blockDim.x + threadIdx.x;
    if (tid >= BATCH * D_INNER) return;
    const int b = tid >> 10;
    const int d = tid & (D_INNER - 1);

    float Aneg[D_STATE];
#pragma unroll
    for (int s = 0; s < D_STATE; ++s) Aneg[s] = -__expf(a_log[d * D_STATE + s]);
    float h[D_STATE];
#pragma unroll
    for (int s = 0; s < D_STATE; ++s) h[s] = 0.0f;

    for (int t = 0; t < SEQ; ++t) {
        const int row = b * SEQ + t;
        const float dtv = dt[(size_t)row * D_INNER + d];
        const float dtx = dtv * xc[(size_t)row * D_INNER + d];
        const float* Bp = xdb + (size_t)row * 64 + DT_RANK;
        const float* Cp = Bp + D_STATE;
        float y = 0.0f;
#pragma unroll
        for (int s = 0; s < D_STATE; ++s) {
            h[s] = h[s] * __expf(dtv * Aneg[s]) + dtx * Bp[s];
            y += h[s] * Cp[s];
        }
        yscan[(size_t)row * D_INNER + d] = y;
    }
}

// gated = (yscan + xc * D[d]) * silu(z);  z = xz[:, D_INNER + d]
__global__ void mixer_gate_kernel(const float* __restrict__ yscan,
                                  const float* __restrict__ xc,
                                  const float* __restrict__ xz,
                                  const float* __restrict__ Dv,
                                  float* __restrict__ out) {
    int i = blockIdx.x * blockDim.x + threadIdx.x;
    if (i >= MTOK * D_INNER) return;
    const int d = i & (D_INNER - 1);
    const int row = i >> 10;
    const float z = xz[(size_t)row * (2 * D_INNER) + D_INNER + d];
    out[i] = (yscan[i] + xc[i] * Dv[d]) * siluf(z);
}

// g2[:, j] = h1[:, j] * silu(h1[:, D_MLP + j])
__global__ void mlp_gate_kernel(const float* __restrict__ h1,
                                float* __restrict__ g2) {
    int i = blockIdx.x * blockDim.x + threadIdx.x;
    if (i >= MTOK * D_MLP) return;
    const int j = i & (D_MLP - 1);
    const int row = i >> 11;
    const float* hr = h1 + (size_t)row * (2 * D_MLP);
    g2[i] = hr[j] * siluf(hr[D_MLP + j]);
}

// pooled[b, c] = mean over t of x[(b*SEQ+t), c]
__global__ void pool_kernel(const float* __restrict__ x, float* __restrict__ pooled) {
    int i = blockIdx.x * blockDim.x + threadIdx.x;
    if (i >= BATCH * D_MODEL) return;
    const int b = i >> 9;
    const int c = i & (D_MODEL - 1);
    const float* xr = x + (size_t)b * SEQ * D_MODEL + c;
    float s = 0.0f;
    for (int t = 0; t < SEQ; ++t) s += xr[(size_t)t * D_MODEL];
    pooled[i] = s * (1.0f / SEQ);
}

__global__ void head_kernel(const float* __restrict__ pooled,
                            const float* __restrict__ hw,
                            const float* __restrict__ hb,
                            float* __restrict__ out) {
    int i = blockIdx.x * blockDim.x + threadIdx.x;
    if (i >= BATCH * N_CLASSES) return;
    const int b = i / N_CLASSES;
    const int c = i % N_CLASSES;
    const float* p = pooled + b * D_MODEL;
    const float* w = hw + c * D_MODEL;
    float acc = hb[c];
    for (int k = 0; k < D_MODEL; ++k) acc += p[k] * w[k];
    out[i] = acc;
}

// ------------------------------- host side ---------------------------------

static inline void launch_gemm(const float* A, int lda, const float* W, int ldw,
                               const float* bias, float* C, int ldc,
                               int M, int N, int K, int act, hipStream_t s) {
    dim3 grid(M / BM, N / BN);
    dim3 block(256);
    if (bias) {
        if (act == 1)
            wmma_gemm_kernel<1, true><<<grid, block, 0, s>>>(A, lda, W, ldw, bias, C, ldc, K);
        else
            wmma_gemm_kernel<0, true><<<grid, block, 0, s>>>(A, lda, W, ldw, bias, C, ldc, K);
    } else {
        wmma_gemm_kernel<0, false><<<grid, block, 0, s>>>(A, lda, W, ldw, nullptr, C, ldc, K);
    }
}

static inline dim3 ew_grid(int n) { return dim3((n + 255) / 256); }

extern "C" void kernel_launch(void* const* d_in, const int* in_sizes, int n_in,
                              void* d_out, int out_size, void* d_ws, size_t ws_size,
                              hipStream_t stream) {
    (void)in_sizes; (void)n_in; (void)out_size; (void)ws_size;

    const float* x         = (const float*)d_in[0];   // [4,512,128]
    const float* embed_w   = (const float*)d_in[1];   // [512,128]
    const float* embed_b   = (const float*)d_in[2];   // [512]
    const float* in_proj_w = (const float*)d_in[3];   // [12,2048,512]
    const float* conv_w    = (const float*)d_in[4];   // [12,1024,4]
    const float* conv_b    = (const float*)d_in[5];   // [12,1024]
    const float* x_proj_w  = (const float*)d_in[6];   // [12,64,1024]
    const float* dt_proj_w = (const float*)d_in[7];   // [12,1024,32]
    const float* dt_proj_b = (const float*)d_in[8];   // [12,1024]
    const float* A_log     = (const float*)d_in[9];   // [12,1024,16]
    const float* Dv        = (const float*)d_in[10];  // [12,1024]
    const float* out_proj_w= (const float*)d_in[11];  // [12,512,1024]
    const float* norm1_w   = (const float*)d_in[12];  // [12,512]
    const float* norm2_w   = (const float*)d_in[13];  // [12,512]
    const float* fc1_w     = (const float*)d_in[14];  // [12,4096,512]
    const float* fc2_w     = (const float*)d_in[15];  // [12,512,2048]
    const float* norm_f_w  = (const float*)d_in[16];  // [512]
    const float* head_w    = (const float*)d_in[17];  // [50,512]
    const float* head_b    = (const float*)d_in[18];  // [50]
    float* out = (float*)d_out;                       // [4,50] f32

    // scratch layout (floats), ~97 MB total
    float* ws = (float*)d_ws;
    float* hidden = ws;  ws += (size_t)MTOK * D_MODEL;
    float* resid  = ws;  ws += (size_t)MTOK * D_MODEL;
    float* xnorm  = ws;  ws += (size_t)MTOK * D_MODEL;
    float* xz     = ws;  ws += (size_t)MTOK * 2 * D_INNER;   // reused as MLP gate buf
    float* xc     = ws;  ws += (size_t)MTOK * D_INNER;
    float* xdb    = ws;  ws += (size_t)MTOK * 64;
    float* dtbuf  = ws;  ws += (size_t)MTOK * D_INNER;
    float* yscan  = ws;  ws += (size_t)MTOK * D_INNER;
    float* gated  = ws;  ws += (size_t)MTOK * D_INNER;
    float* h1     = ws;  ws += (size_t)MTOK * 2 * D_MLP;
    float* pooled = ws;  ws += (size_t)BATCH * D_MODEL;

    const int nHid = MTOK * D_MODEL;

    // residual = 0; hidden = x @ embed_w.T + embed_b
    zero_kernel<<<ew_grid(nHid), 256, 0, stream>>>(resid, nHid);
    launch_gemm(x, N_MELS, embed_w, N_MELS, embed_b, hidden, D_MODEL,
                MTOK, D_MODEL, N_MELS, 0, stream);

    for (int i = 0; i < N_LAYER; ++i) {
        const float* inw    = in_proj_w + (size_t)i * 2 * D_INNER * D_MODEL;
        const float* cw     = conv_w    + (size_t)i * D_INNER * D_CONV;
        const float* cb     = conv_b    + (size_t)i * D_INNER;
        const float* xpw    = x_proj_w  + (size_t)i * 64 * D_INNER;
        const float* dtw    = dt_proj_w + (size_t)i * D_INNER * DT_RANK;
        const float* dtbias = dt_proj_b + (size_t)i * D_INNER;
        const float* alog   = A_log     + (size_t)i * D_INNER * D_STATE;
        const float* dsk    = Dv        + (size_t)i * D_INNER;
        const float* ow     = out_proj_w+ (size_t)i * D_MODEL * D_INNER;
        const float* n1     = norm1_w   + (size_t)i * D_MODEL;
        const float* n2     = norm2_w   + (size_t)i * D_MODEL;
        const float* f1     = fc1_w     + (size_t)i * 2 * D_MLP * D_MODEL;
        const float* f2     = fc2_w     + (size_t)i * D_MODEL * D_MLP;

        // ---- mixer ----
        add_kernel<<<ew_grid(nHid), 256, 0, stream>>>(resid, hidden, nHid);
        rmsnorm_kernel<<<MTOK / 8, 256, 0, stream>>>(resid, n1, xnorm, MTOK);

        launch_gemm(xnorm, D_MODEL, inw, D_MODEL, nullptr, xz, 2 * D_INNER,
                    MTOK, 2 * D_INNER, D_MODEL, 0, stream);

        conv_silu_kernel<<<ew_grid(MTOK * D_INNER), 256, 0, stream>>>(xz, cw, cb, xc);

        launch_gemm(xc, D_INNER, xpw, D_INNER, nullptr, xdb, 64,
                    MTOK, 64, D_INNER, 0, stream);

        launch_gemm(xdb, 64, dtw, DT_RANK, dtbias, dtbuf, D_INNER,
                    MTOK, D_INNER, DT_RANK, 1, stream);

        scan_kernel<<<(BATCH * D_INNER) / 256, 256, 0, stream>>>(
            xdb, dtbuf, xc, alog, yscan);

        mixer_gate_kernel<<<ew_grid(MTOK * D_INNER), 256, 0, stream>>>(
            yscan, xc, xz, dsk, gated);

        launch_gemm(gated, D_INNER, ow, D_INNER, nullptr, hidden, D_MODEL,
                    MTOK, D_MODEL, D_INNER, 0, stream);

        // ---- MLP ----
        add_kernel<<<ew_grid(nHid), 256, 0, stream>>>(resid, hidden, nHid);
        rmsnorm_kernel<<<MTOK / 8, 256, 0, stream>>>(resid, n2, xnorm, MTOK);

        launch_gemm(xnorm, D_MODEL, f1, D_MODEL, nullptr, h1, 2 * D_MLP,
                    MTOK, 2 * D_MLP, D_MODEL, 0, stream);

        mlp_gate_kernel<<<ew_grid(MTOK * D_MLP), 256, 0, stream>>>(h1, xz);

        launch_gemm(xz, D_MLP, f2, D_MLP, nullptr, hidden, D_MODEL,
                    MTOK, D_MODEL, D_MLP, 0, stream);
    }

    // final norm + mean pool + head
    add_kernel<<<ew_grid(nHid), 256, 0, stream>>>(resid, hidden, nHid);
    rmsnorm_kernel<<<MTOK / 8, 256, 0, stream>>>(resid, norm_f_w, xnorm, MTOK);
    pool_kernel<<<ew_grid(BATCH * D_MODEL), 256, 0, stream>>>(xnorm, pooled);
    head_kernel<<<1, 256, 0, stream>>>(pooled, head_w, head_b, out);
}
